// GraphTransformerLayer_70772471103744
// MI455X (gfx1250) — matrix operand order repaired and checked by reading, not compiled, not run on hardware
//
#include <hip/hip_runtime.h>
#include <math.h>

// ---------------------------------------------------------------------------
// Types / constants
// ---------------------------------------------------------------------------
typedef unsigned short bf16_t;
typedef __attribute__((ext_vector_type(16))) __bf16 v16bf;
typedef __attribute__((ext_vector_type(8)))  float  v8f;
typedef __attribute__((ext_vector_type(4)))  float  v4f;
typedef __attribute__((ext_vector_type(4)))  unsigned int v4u;
typedef __attribute__((ext_vector_type(8)))  int    v8i;
typedef __attribute__((ext_vector_type(4)))  int    v4i;

#define D_MODEL 1024
#define NSEQ    2048
#define NHEAD   16
#define DHEAD   64
#define SCALE_QK 0.125f   /* 1/sqrt(64) */

__device__ __forceinline__ bf16_t f2bf(float f) {
  // round-to-nearest-even fp32 -> bf16
  unsigned int u = __float_as_uint(f);
  unsigned int r = ((u >> 16) & 1u) + 0x7FFFu;
  return (bf16_t)((u + r) >> 16);
}

union frag_u { v16bf v; v4f q[2]; };

// A-matrix fragment (16xK tile, one 32-wide K window starting at `base`, which
// points at this lane's row). ISA 16-bit A layout: lanes 0-15 and 16-31 both
// hold rows M=0..15; group g elems = K {g*8..g*8+7, 16+g*8..16+g*8+7}.
__device__ __forceinline__ v16bf load_fragA(const bf16_t* base, int g) {
  frag_u u;
  u.q[0] = *(const v4f*)(base + (g << 3));
  u.q[1] = *(const v4f*)(base + 16 + (g << 3));
  return u.v;
}

// B-matrix fragment (32x16 tile; `base` points at this lane's column, laid out
// contiguously along K). Same K order as A: group g elems =
// K {g*8..g*8+7, 16+g*8..16+g*8+7}.
__device__ __forceinline__ v16bf load_fragB(const bf16_t* base, int g) {
  frag_u u;
  u.q[0] = *(const v4f*)(base + (g << 3));
  u.q[1] = *(const v4f*)(base + 16 + (g << 3));
  return u.v;
}

__device__ __forceinline__ v8f wmma_bf16(v16bf a, v16bf b, v8f c) {
  // (neg_a, A, neg_b, B, c_mod, C, reuse_a, reuse_b)
  return __builtin_amdgcn_wmma_f32_16x16x32_bf16(false, a, false, b, (short)0,
                                                 c, false, false);
}

// ---------------------------------------------------------------------------
// Tensor Data Mover: DMA one 64(N) x 32(K) bf16 tile (row stride K elements)
// from global into LDS at byte offset `lds_off`, contiguous 64B rows.
// D# packing per CDNA5 ISA 8.3/8.4: count=1, data_size=1(2B), tile_dim0=32,
// tile_dim1=64, tensor_dim0=K, tensor_dim1=64, tensor_dim0_stride=K, type=2.
// Issued by one wave; completion tracked with TENSORcnt.
// This toolchain exposes the 6-arg builtin (g0, g1, g2, g3, g4, cpol).
// ---------------------------------------------------------------------------
__device__ __forceinline__ void tdm_load_tile_64x32(unsigned lds_off,
                                                    const bf16_t* gsrc,
                                                    int K) {
  const unsigned long long ga = (unsigned long long)(size_t)gsrc;
  v4u g0;
  g0[0] = 1u;                       // count=1, is_restore=0, gather off
  g0[1] = lds_off;                  // lds_addr (bytes)
  g0[2] = (unsigned)ga;             // global_addr[31:0]
  g0[3] = ((unsigned)(ga >> 32) & 0x01FFFFFFu) | 0x80000000u;  // [56:32]|type=2
  v8i g1;
  g1[0] = 0x10000;                          // data_size=1 (2 bytes)
  g1[1] = (int)(((unsigned)K & 0xFFFFu) << 16);   // tensor_dim0[15:0]
  g1[2] = (int)(((unsigned)K >> 16) | (64u << 16)); // tdim0[31:16]|tdim1[15:0]
  g1[3] = (int)(32u << 16);                 // tile_dim0 = 32
  g1[4] = 64;                               // tile_dim1 = 64
  g1[5] = K;                                // tensor_dim0_stride[31:0]
  g1[6] = 0;
  g1[7] = 0;
  const v4i z4 = {0, 0, 0, 0};              // groups 2/3 unused (2-D tensor)
  const v8i z8 = {0, 0, 0, 0, 0, 0, 0, 0};  // trailing group unused
  __builtin_amdgcn_tensor_load_to_lds(g0, g1, z4, z4, z8, 0);
}

// ---------------------------------------------------------------------------
// fp32 [K][N] weight  ->  bf16 transposed [N][K]  (one-time, ~25MB total)
// ---------------------------------------------------------------------------
__global__ __launch_bounds__(256) void convert_transpose_bf16(
    const float* __restrict__ W, bf16_t* __restrict__ Wt, int K, int N) {
  const int n = blockIdx.y;
  const int k = blockIdx.x * 256 + threadIdx.x;
  Wt[(size_t)n * K + k] = f2bf(W[(size_t)k * N + n]);
}

// ---------------------------------------------------------------------------
// LayerNorm: fp32 [rows][1024] -> bf16 normalized output
// ---------------------------------------------------------------------------
__global__ __launch_bounds__(256) void layernorm_to_bf16(
    const float* __restrict__ x, const float* __restrict__ gw,
    const float* __restrict__ bw, bf16_t* __restrict__ out) {
  const int row = blockIdx.x;
  const float* xr = x + (size_t)row * D_MODEL;
  float v[4], s = 0.f, ss = 0.f;
#pragma unroll
  for (int i = 0; i < 4; ++i) {
    v[i] = xr[threadIdx.x + 256 * i];
    s += v[i];
    ss += v[i] * v[i];
  }
#pragma unroll
  for (int off = 1; off < 32; off <<= 1) {
    s  += __shfl_xor(s,  off, 32);
    ss += __shfl_xor(ss, off, 32);
  }
  __shared__ float red[2][8];
  const int w = threadIdx.x >> 5, lane = threadIdx.x & 31;
  if (lane == 0) { red[0][w] = s; red[1][w] = ss; }
  __syncthreads();
  s = 0.f; ss = 0.f;
#pragma unroll
  for (int i = 0; i < 8; ++i) { s += red[0][i]; ss += red[1][i]; }
  const float mu   = s * (1.f / D_MODEL);
  const float var  = ss * (1.f / D_MODEL) - mu * mu;
  const float rstd = rsqrtf(var + 1e-5f);
#pragma unroll
  for (int i = 0; i < 4; ++i) {
    const int c = threadIdx.x + 256 * i;
    out[(size_t)row * D_MODEL + c] = f2bf((v[i] - mu) * rstd * gw[c] + bw[c]);
  }
}

// ---------------------------------------------------------------------------
// Generic bf16 WMMA GEMM: C[M][N] = A[M][K] * Bt[N][K]^T, epilogue functor.
// 256 threads = 8 waves; block tile 128(M) x 64(N); K-step 32.
// B tiles are DMA'd into double-buffered LDS by the Tensor Data Mover
// (issued by wave 0, tracked with TENSORcnt) while all 8 waves run WMMAs.
// ---------------------------------------------------------------------------
template <typename Epi>
__global__ __launch_bounds__(256) void gemm_bf16_wmma(
    const bf16_t* __restrict__ A, const bf16_t* __restrict__ Bt, int K,
    Epi epi) {
  __shared__ __align__(16) bf16_t Bs[2][64 * 32];
  const int w = threadIdx.x >> 5;
  const int lane = threadIdx.x & 31;
  const int g = lane >> 4;
  const int j = lane & 15;
  const int n0 = blockIdx.x * 64;
  const int m0 = blockIdx.y * 128;
  const bf16_t* Arow = A + (size_t)(m0 + w * 16 + j) * K;
  const bf16_t* Bblk = Bt + (size_t)n0 * K;
  // LDS byte offsets of the two B-tile buffers (low 32 bits of generic addr)
  const unsigned ldsBuf0 = (unsigned)(size_t)(void*)&Bs[0][0];
  const unsigned ldsBuf1 = ldsBuf0 + 64 * 32 * 2;

  // prologue: DMA tile for k0=0 into buffer 0
  if (w == 0) tdm_load_tile_64x32(ldsBuf0, Bblk, K);

  v8f acc[4] = {};
  int buf = 0;
  for (int k0 = 0; k0 < K; k0 += 32, buf ^= 1) {
    if (k0 + 32 < K) {  // DMA next tile into the other buffer
      if (w == 0)
        tdm_load_tile_64x32(buf ? ldsBuf0 : ldsBuf1, Bblk + k0 + 32, K);
      __builtin_amdgcn_s_wait_tensorcnt(1);  // current tile complete
    } else {
      __builtin_amdgcn_s_wait_tensorcnt(0);
    }
    __syncthreads();  // current tile visible to all 8 waves
    __builtin_prefetch(Arow + k0 + 64, 0, 3);
    const v16bf a = load_fragA(Arow + k0, g);
    const bf16_t* Bcur = &Bs[buf][0];
#pragma unroll
    for (int t = 0; t < 4; ++t) {
      const v16bf b = load_fragB(Bcur + (16 * t + j) * 32, g);
      acc[t] = wmma_bf16(a, b, acc[t]);
    }
    __syncthreads();  // done reading before this buffer is DMA-overwritten
  }
  // C layout: VGPR r holds row m0+w*16+g*8+r, lane j holds column n0+16t+j
#pragma unroll
  for (int t = 0; t < 4; ++t)
#pragma unroll
    for (int r = 0; r < 8; ++r)
      epi(acc[t][r], m0 + w * 16 + g * 8 + r, n0 + 16 * t + j);
}

// ---------------------------------------------------------------------------
// GEMM epilogues
// ---------------------------------------------------------------------------
struct EpiQKV {  // bias add + scatter to Q[bh][n][dh], K[bh][n][dh], Vt[bh][dh][n]
  const float* bias; bf16_t* Qb; bf16_t* Kb; bf16_t* Vt;
  __device__ void operator()(float a, int m, int n) const {
    const float v = a + bias[n];
    const int b = m >> 11, tok = m & 2047;
    const int s = n >> 10, h = (n >> 6) & 15, dh = n & 63;
    const int bh = b * NHEAD + h;
    if (s == 0)      Qb[((size_t)bh * NSEQ + tok) * DHEAD + dh] = f2bf(v);
    else if (s == 1) Kb[((size_t)bh * NSEQ + tok) * DHEAD + dh] = f2bf(v);
    else             Vt[((size_t)bh * DHEAD + dh) * NSEQ + tok] = f2bf(v);
  }
};
struct EpiOutProj {  // + out_b + residual(x) -> x1 fp32
  const float* bias; const float* resid; float* x1;
  __device__ void operator()(float a, int m, int n) const {
    x1[(size_t)m * D_MODEL + n] =
        a + bias[n] + resid[(size_t)m * D_MODEL + n];
  }
};
struct EpiGelu {  // + ff1_b, exact GELU -> bf16 activation
  const float* bias; bf16_t* o;
  __device__ void operator()(float a, int m, int n) const {
    const float v = a + bias[n];
    o[(size_t)m * (4 * D_MODEL) + n] =
        f2bf(0.5f * v * (1.f + erff(v * 0.70710678118654752f)));
  }
};
struct EpiFF2 {  // + ff2_b + residual(x1) -> final output fp32
  const float* bias; const float* x1; float* o;
  __device__ void operator()(float a, int m, int n) const {
    o[(size_t)m * D_MODEL + n] = a + bias[n] + x1[(size_t)m * D_MODEL + n];
  }
};

// ---------------------------------------------------------------------------
// Flash attention with WMMA. 128 threads = 4 waves; 16 queries per wave;
// key blocks of 32; online softmax; P re-shaped via wave-private LDS tile;
// V consumed pre-transposed so P*V B-fragments are contiguous.
// ---------------------------------------------------------------------------
__global__ __launch_bounds__(128) void attn_flash_wmma(
    const bf16_t* __restrict__ Qb, const bf16_t* __restrict__ Kb,
    const bf16_t* __restrict__ Vt, const float* __restrict__ adj,
    const float* __restrict__ amask, const float* __restrict__ sbias,
    bf16_t* __restrict__ attnout) {
  __shared__ __align__(16) bf16_t Plds[4][16 * 32];
  const int w = threadIdx.x >> 5;
  const int lane = threadIdx.x & 31;
  const int g = lane >> 4;
  const int j = lane & 15;
  const int bh = blockIdx.y;
  const int b = bh >> 4;
  const int h = bh & 15;
  const int q0 = blockIdx.x * 64 + w * 16;
  const float sb = sbias[0];

  const bf16_t* Qrow = Qb + ((size_t)bh * NSEQ + q0 + j) * DHEAD;
  const v16bf aq0 = load_fragA(Qrow, g);        // dh 0..31
  const v16bf aq1 = load_fragA(Qrow + 32, g);   // dh 32..63

  v8f O[4] = {};
  float mrow[8], lrow[8];
#pragma unroll
  for (int r = 0; r < 8; ++r) { mrow[r] = -1e30f; lrow[r] = 0.f; }

  const float* adjb = adj + (size_t)b * NSEQ * NSEQ;
  const float* mskb = amask + (size_t)b * NSEQ * NSEQ;
  const bf16_t* Kbase = Kb + (size_t)bh * NSEQ * DHEAD;
  const bf16_t* Vbase = Vt + (size_t)bh * DHEAD * NSEQ;
  bf16_t* Pw = &Plds[w][0];

  for (int k0 = 0; k0 < NSEQ; k0 += 32) {
    // ---- S = Q*K^T for 16(q) x 32(k), K-dim = Dh = 64 (2 WMMAs per tile)
    v8f S[2] = {};
#pragma unroll
    for (int t = 0; t < 2; ++t) {
      const bf16_t* Krow = Kbase + (size_t)(k0 + 16 * t + j) * DHEAD;
      S[t] = wmma_bf16(aq0, load_fragB(Krow, g), S[t]);
      S[t] = wmma_bf16(aq1, load_fragB(Krow + 32, g), S[t]);
    }
    // ---- scale + struct_bias*adj + mask, online softmax (rows across VGPRs)
#pragma unroll
    for (int r = 0; r < 8; ++r) {
      const int qi = q0 + g * 8 + r;
      const size_t ro = (size_t)qi * NSEQ + k0 + j;
      float s0 = S[0][r] * SCALE_QK + sb * adjb[ro]      + mskb[ro];
      float s1 = S[1][r] * SCALE_QK + sb * adjb[ro + 16] + mskb[ro + 16];
      float mx = fmaxf(s0, s1);
#pragma unroll
      for (int off = 1; off < 16; off <<= 1)   // reduce over the 16 columns
        mx = fmaxf(mx, __shfl_xor(mx, off, 32));
      const float mnew  = fmaxf(mrow[r], mx);
      const float alpha = __expf(mrow[r] - mnew);
      const float p0 = __expf(s0 - mnew);
      const float p1 = __expf(s1 - mnew);
      float ps = p0 + p1;
#pragma unroll
      for (int off = 1; off < 16; off <<= 1)
        ps += __shfl_xor(ps, off, 32);
      lrow[r] = lrow[r] * alpha + ps;
      mrow[r] = mnew;
#pragma unroll
      for (int t = 0; t < 4; ++t) O[t][r] *= alpha;   // rescale accumulator
      // stash P tile (16x32) bf16 in wave-private LDS, natural (row,key) order
      Pw[(g * 8 + r) * 32 + j]      = f2bf(p0);
      Pw[(g * 8 + r) * 32 + 16 + j] = f2bf(p1);
    }
    __builtin_amdgcn_wave_barrier();  // keep ds writes before ds reads
    // ---- O += P * V  (A-frag from LDS, B-frags contiguous rows of Vt)
    const v16bf ap = load_fragA(&Pw[j * 32], g);
#pragma unroll
    for (int t = 0; t < 4; ++t) {
      const bf16_t* Vrow = Vbase + (size_t)(16 * t + j) * NSEQ + k0;
      O[t] = wmma_bf16(ap, load_fragB(Vrow, g), O[t]);
    }
  }
  // ---- normalize and emit bf16 rows of the attention output matrix
  bf16_t* orow =
      attnout + (size_t)(b * NSEQ + q0 + g * 8) * D_MODEL + h * DHEAD;
#pragma unroll
  for (int r = 0; r < 8; ++r) {
    const float inv = 1.0f / lrow[r];
#pragma unroll
    for (int t = 0; t < 4; ++t)
      orow[(size_t)r * D_MODEL + 16 * t + j] = f2bf(O[t][r] * inv);
  }
}

// ---------------------------------------------------------------------------
// Host orchestration
// ---------------------------------------------------------------------------
extern "C" void kernel_launch(void* const* d_in, const int* in_sizes, int n_in,
                              void* d_out, int out_size, void* d_ws,
                              size_t ws_size, hipStream_t stream) {
  (void)in_sizes; (void)n_in; (void)out_size; (void)ws_size;
  const float* x      = (const float*)d_in[0];
  const float* adj    = (const float*)d_in[1];
  const float* amask  = (const float*)d_in[2];
  const float* qkv_w  = (const float*)d_in[3];
  const float* qkv_b  = (const float*)d_in[4];
  const float* out_w  = (const float*)d_in[5];
  const float* out_b  = (const float*)d_in[6];
  const float* ln1_g  = (const float*)d_in[7];
  const float* ln1_b  = (const float*)d_in[8];
  const float* ln2_g  = (const float*)d_in[9];
  const float* ln2_b  = (const float*)d_in[10];
  const float* ff1_w  = (const float*)d_in[11];
  const float* ff1_b  = (const float*)d_in[12];
  const float* ff2_w  = (const float*)d_in[13];
  const float* ff2_b  = (const float*)d_in[14];
  const float* sbias  = (const float*)d_in[15];
  float* out = (float*)d_out;

  const int M = 2 * NSEQ;  // 4096 tokens
  char* ws = (char*)d_ws;
  size_t off = 0;
  auto alloc = [&](size_t bytes) -> void* {
    void* p = ws + off;
    off = (off + bytes + 255) & ~(size_t)255;
    return p;
  };
  bf16_t* qkvWt  = (bf16_t*)alloc((size_t)3072 * 1024 * 2);
  bf16_t* outWt  = (bf16_t*)alloc((size_t)1024 * 1024 * 2);
  bf16_t* ff1Wt  = (bf16_t*)alloc((size_t)4096 * 1024 * 2);
  bf16_t* ff2Wt  = (bf16_t*)alloc((size_t)1024 * 4096 * 2);
  bf16_t* h1     = (bf16_t*)alloc((size_t)M * 1024 * 2);
  bf16_t* Qb     = (bf16_t*)alloc((size_t)32 * NSEQ * DHEAD * 2);
  bf16_t* Kb     = (bf16_t*)alloc((size_t)32 * NSEQ * DHEAD * 2);
  bf16_t* Vtr    = (bf16_t*)alloc((size_t)32 * NSEQ * DHEAD * 2);
  bf16_t* attno  = (bf16_t*)alloc((size_t)M * 1024 * 2);
  float*  x1     = (float*)alloc((size_t)M * 1024 * 4);
  bf16_t* h2     = (bf16_t*)alloc((size_t)M * 1024 * 2);
  bf16_t* ffact  = (bf16_t*)alloc((size_t)M * 4096 * 2);

  // 1) weight convert + transpose (W[K][N] -> bf16 Wt[N][K])
  convert_transpose_bf16<<<dim3(4, 3072), 256, 0, stream>>>(qkv_w, qkvWt, 1024, 3072);
  convert_transpose_bf16<<<dim3(4, 1024), 256, 0, stream>>>(out_w, outWt, 1024, 1024);
  convert_transpose_bf16<<<dim3(4, 4096), 256, 0, stream>>>(ff1_w, ff1Wt, 1024, 4096);
  convert_transpose_bf16<<<dim3(16, 1024), 256, 0, stream>>>(ff2_w, ff2Wt, 4096, 1024);

  // 2) LN1 -> bf16
  layernorm_to_bf16<<<M, 256, 0, stream>>>(x, ln1_g, ln1_b, h1);
  // 3) QKV GEMM, epilogue scatters bf16 Q/K + pre-transposed V
  gemm_bf16_wmma<EpiQKV><<<dim3(48, 32), 256, 0, stream>>>(
      h1, qkvWt, 1024, EpiQKV{qkv_b, Qb, Kb, Vtr});
  // 4) flash attention (16 heads x 2 batches, 64 queries per block)
  attn_flash_wmma<<<dim3(32, 32), 128, 0, stream>>>(Qb, Kb, Vtr, adj, amask,
                                                    sbias, attno);
  // 5) out projection + residual -> x1
  gemm_bf16_wmma<EpiOutProj><<<dim3(16, 32), 256, 0, stream>>>(
      attno, outWt, 1024, EpiOutProj{out_b, x, x1});
  // 6) LN2 -> bf16
  layernorm_to_bf16<<<M, 256, 0, stream>>>(x1, ln2_g, ln2_b, h2);
  // 7) FF1 + GELU -> bf16
  gemm_bf16_wmma<EpiGelu><<<dim3(64, 32), 256, 0, stream>>>(
      h2, ff1Wt, 1024, EpiGelu{ff1_b, ffact});
  // 8) FF2 + residual -> output
  gemm_bf16_wmma<EpiFF2><<<dim3(16, 32), 256, 0, stream>>>(
      ffact, ff2Wt, 4096, EpiFF2{ff2_b, x1, out});
}
